// SurfaceCodeGNN_8349416423591
// MI455X (gfx1250) — matrix operand order repaired and checked by reading, not compiled, not run on hardware
//
#include <hip/hip_runtime.h>
#include <hip/hip_bf16.h>
#include <hip/hip_fp16.h>
#include <math.h>

// ---------------- problem constants (match reference) ----------------
#define SC_S 8
#define SC_Q 9
#define SC_F 64
#define SC_H 256
#define SC_ITERS 3
#define SC_B 65536
#define SC_N (SC_B * SC_S)        // 524288 nodes
#define SC_E 2097152              // edges

typedef __attribute__((ext_vector_type(16))) _Float16 v16h;
typedef __attribute__((ext_vector_type(8)))  _Float16 v8h;
typedef __attribute__((ext_vector_type(8)))  short    v8s;
typedef __attribute__((ext_vector_type(8)))  float    v8f;

__device__ __forceinline__ v8f wmma_f32_16x16x32(v16h a, v16h b, v8f c) {
  return __builtin_amdgcn_wmma_f32_16x16x32_f16(
      /*neg_a=*/false, a, /*neg_b=*/false, b,
      /*c_mod=*/(short)0, c, /*reuse_a=*/false, /*reuse_b=*/false);
}

// Packed f16 relu via sign-mask (no fp canonicalize): m = x>>15; x & ~m.
__device__ __forceinline__ v8h relu_pk8(v8h x) {
  v8s xi;
  __builtin_memcpy(&xi, &x, sizeof(xi));
  const v8s m = xi >> 15;          // all-ones where negative
  xi = xi & ~m;
  v8h r;
  __builtin_memcpy(&r, &xi, sizeof(r));
  return r;
}

// =====================================================================
// Weight pre-pack: W[K, Nact] (f32, row-major) -> P, f16, per-lane WMMA
// B layout: P[((kt*NT)+nt)*512 + lane*16 + e];
//   k = kt*32 + 8*(lane>>4) + (e&7) + 16*(e>>3), col = nt*16 + (lane&15); col>=Nact -> 0.
// =====================================================================
__global__ void pack_w_kernel(const float* __restrict__ W, _Float16* __restrict__ P,
                              int K, int Nact, int NT) {
  const int total = (K >> 5) * NT * 512;
  const int i = blockIdx.x * blockDim.x + threadIdx.x;
  if (i >= total) return;
  const int e    = i & 15;
  const int lane = (i >> 4) & 31;
  const int tile = i >> 9;               // kt*NT + nt
  const int nt   = tile % NT;
  const int kt   = tile / NT;
  const int g    = lane >> 4;
  const int m16  = lane & 15;
  const int k    = kt * 32 + 8 * g + (e & 7) + 16 * (e >> 3);
  const int col  = nt * 16 + m16;
  const float v  = (col < Nact) ? W[(size_t)k * Nact + col] : 0.0f;
  P[i] = (_Float16)v;
}

// packed-B tile load; NTstride = NT used at pack time, nt = absolute n-tile
__device__ __forceinline__ v16h load_bp(const _Float16* __restrict__ Bp,
                                        int kk, int nt, int NTstride, int lane) {
  const size_t off = ((size_t)(kk >> 5) * NTstride + nt) * 512 + (size_t)lane * 16;
  return *(const v16h*)(Bp + off);
}

// Direct per-lane A operand from row-major f16 [*, K]: two aligned 16B runs.
// g=0: K = kk+0..7 and kk+16..23 ; g=1: K = kk+8..15 and kk+24..31.
template <bool RELU>
__device__ __forceinline__ v16h load_a_f16(const _Float16* __restrict__ A,
                                           int row, int K, int kk, int g) {
  const _Float16* ap = A + (size_t)row * K + kk + 8 * g;
  v8h alo = *(const v8h*)ap;
  v8h ahi = *(const v8h*)(ap + 16);
  if (RELU) { alo = relu_pk8(alo); ahi = relu_pk8(ahi); }
  v16h a;
#pragma unroll
  for (int e = 0; e < 8; ++e) { a[e] = alo[e]; a[e + 8] = ahi[e]; }
  return a;
}

// =====================================================================
// Generic GEMM: C[M, cols nt0*16 ..] = act(A[M,K] @ W + bias).
// A f16 row-major; W pre-packed f16 with stride NTstride; C f16 or f32.
// Block = 256 threads = 8 waves; wave owns 16 rows x (NTILES*16) cols.
// ACT: 0=none, 1=relu, 2=sigmoid.  NGUARD: guard output cols vs Nout.
// =====================================================================
template <int NTILES, int ACT, bool NGUARD, bool OUTF16>
__global__ __launch_bounds__(256) void gemm_wmma_kernel(
    const _Float16* __restrict__ A, const _Float16* __restrict__ Bp,
    const float* __restrict__ bias, void* __restrict__ Cv,
    int M, int K, int Nout, int nt0, int NTstride) {
  const int tid  = threadIdx.x;
  const int wave = tid >> 5;
  const int lane = tid & 31;
  const int g    = lane >> 4;
  const int m16  = lane & 15;
  const int myrow = (blockIdx.x * 8 + wave) * 16;

  v8f acc[NTILES];
#pragma unroll
  for (int t = 0; t < NTILES; ++t) acc[t] = v8f{0, 0, 0, 0, 0, 0, 0, 0};

  for (int kk = 0; kk < K; kk += 32) {
    const v16h a = load_a_f16<false>(A, myrow + m16, K, kk, g);
    v16h breg[NTILES];
#pragma unroll
    for (int t = 0; t < NTILES; ++t)
      breg[t] = load_bp(Bp, kk, nt0 + t, NTstride, lane);
#pragma unroll
    for (int t = 0; t < NTILES; ++t)
      acc[t] = wmma_f32_16x16x32(a, breg[t], acc[t]);
  }

#pragma unroll
  for (int t = 0; t < NTILES; ++t) {
    const int col = (nt0 + t) * 16 + m16;
    if (NGUARD && col >= Nout) continue;
    const float bv = bias ? bias[col] : 0.0f;
#pragma unroll
    for (int r = 0; r < 8; ++r) {
      const int mrow = myrow + r + 8 * g;
      float v = acc[t][r] + bv;
      if (ACT == 1) v = fmaxf(v, 0.0f);
      else if (ACT == 2) v = 1.0f / (1.0f + __expf(-v));
      if (OUTF16) ((_Float16*)Cv)[(size_t)mrow * Nout + col] = (_Float16)v;
      else        ((float*)Cv)[(size_t)mrow * Nout + col] = v;
    }
  }
}

// =====================================================================
// Fused refine MLP with folded pre-ReLU (h f16 in/out, pre-relu in):
//   hr = relu(h);  r = relu(hr@W1+b1)@W2 + b2;  h = relu(hr + r)
// Stage 1 = two sequential NT=8 passes (unroll disabled so the compiler
// cannot CSE-merge them back into one NT=16 high-pressure region).
// Hidden [16,256] staged as f16 in LDS per wave.
// =====================================================================
__global__ __launch_bounds__(128) void refine_fused_kernel(
    _Float16* __restrict__ h,               // [SC_N, 64] f16 in/out
    const _Float16* __restrict__ W1p,       // packed [64,256], NT=16
    const float* __restrict__ b1,           // [256]
    const _Float16* __restrict__ W2p,       // packed [256,64], NT=4
    const float* __restrict__ b2) {         // [64]
  __shared__ _Float16 sT[4][16][SC_H + 8];  // row = 264 f16 = 528B = 33*16

  const int tid  = threadIdx.x;
  const int wave = tid >> 5;
  const int lane = tid & 31;
  const int g    = lane >> 4;
  const int m16  = lane & 15;
  const int myrow = (blockIdx.x * 4 + wave) * 16;

  // ---------- stage 1: t = relu(relu(h) @ W1 + b1), two NT=8 passes ----------
#pragma unroll 1
  for (int half = 0; half < 2; ++half) {
    v8f acc1[8];
#pragma unroll
    for (int t = 0; t < 8; ++t) acc1[t] = v8f{0, 0, 0, 0, 0, 0, 0, 0};

#pragma unroll
    for (int kk = 0; kk < SC_F; kk += 32) {
      const v16h a = load_a_f16<true>(h, myrow + m16, SC_F, kk, g);
      v16h breg[8];
#pragma unroll
      for (int t = 0; t < 8; ++t)
        breg[t] = load_bp(W1p, kk, half * 8 + t, 16, lane);
#pragma unroll
      for (int t = 0; t < 8; ++t)
        acc1[t] = wmma_f32_16x16x32(a, breg[t], acc1[t]);
    }

    // relu + write hidden half to LDS (f16)
#pragma unroll
    for (int t = 0; t < 8; ++t) {
      const int col = (half * 8 + t) * 16 + m16;
      const float bv = b1[col];
#pragma unroll
      for (int r = 0; r < 8; ++r) {
        const int ml = r + 8 * g;
        sT[wave][ml][col] = (_Float16)fmaxf(acc1[t][r] + bv, 0.0f);
      }
    }
  }
  __syncthreads();

  // ---------- stage 2: r = t @ W2 + b2 ; h = relu(relu(h) + r) ----------
  v8f acc2[4];
#pragma unroll
  for (int t = 0; t < 4; ++t) acc2[t] = v8f{0, 0, 0, 0, 0, 0, 0, 0};

  for (int kk = 0; kk < SC_H; kk += 32) {
    v16h a;
#pragma unroll
    for (int e = 0; e < 16; ++e) {
      const int km = e + 8 * g + ((e >= 8) ? 8 : 0);
      a[e] = sT[wave][m16][kk + km];
    }
    v16h breg[4];
#pragma unroll
    for (int j = 0; j < 4; ++j)
      breg[j] = load_bp(W2p, kk, j, 4, lane);
#pragma unroll
    for (int j = 0; j < 4; ++j)
      acc2[j] = wmma_f32_16x16x32(a, breg[j], acc2[j]);
  }

#pragma unroll
  for (int t = 0; t < 4; ++t) {
    const int col = t * 16 + m16;
    const float bv = b2[col];
#pragma unroll
    for (int r = 0; r < 8; ++r) {
      const int mrow = myrow + r + 8 * g;
      const size_t idx = (size_t)mrow * SC_F + col;
      const float hr = fmaxf((float)h[idx], 0.0f);
      h[idx] = (_Float16)fmaxf(hr + acc2[t][r] + bv, 0.0f);
    }
  }
}

// ---------------- graph / elementwise kernels ----------------
__global__ void init_deg_kernel(float* dinv) {
  size_t i = (size_t)blockIdx.x * blockDim.x + threadIdx.x;
  if (i < SC_N) dinv[i] = 1.0f;  // self loop
}
__global__ void deg_accum_kernel(const long long* __restrict__ dst, float* dinv) {
  size_t e = (size_t)blockIdx.x * blockDim.x + threadIdx.x;
  if (e < SC_E) atomicAdd(&dinv[(size_t)dst[e]], 1.0f);
}
__global__ void deg_finalize_kernel(float* dinv) {
  size_t i = (size_t)blockIdx.x * blockDim.x + threadIdx.x;
  if (i < SC_N) dinv[i] = rsqrtf(dinv[i]);
}
__global__ void edge_norm_kernel(const long long* __restrict__ src,
                                 const long long* __restrict__ dst,
                                 const float* __restrict__ dinv,
                                 float* __restrict__ norm) {
  size_t e = (size_t)blockIdx.x * blockDim.x + threadIdx.x;
  if (e < SC_E) norm[e] = dinv[(size_t)src[e]] * dinv[(size_t)dst[e]];
}
__global__ void embed_kernel(const float* __restrict__ syn,
                             const float* __restrict__ eW,
                             const float* __restrict__ eB,
                             _Float16* __restrict__ h) {
  size_t i = (size_t)blockIdx.x * blockDim.x + threadIdx.x;
  if (i >= (size_t)SC_N * SC_F) return;
  size_t n = i >> 6;
  int f = (int)(i & 63);
  h[i] = (_Float16)(syn[n] * eW[f] + eB[f]);
}
// agg(=h) = xw * dinv^2 (self loop) + bias, packed pairs
__global__ void gcn_selfloop_kernel(const __half2* __restrict__ xw2,
                                    const float* __restrict__ dinv,
                                    const float* __restrict__ gb,
                                    __half2* __restrict__ agg2) {
  size_t i = (size_t)blockIdx.x * blockDim.x + threadIdx.x;
  if (i >= (size_t)SC_N * 32) return;
  size_t n = i >> 5;
  int q = (int)(i & 31);
  const float d = dinv[n];
  const float d2 = d * d;
  const float2 xf = __half22float2(xw2[i]);
  agg2[i] = __floats2half2_rn(xf.x * d2 + gb[2 * q], xf.y * d2 + gb[2 * q + 1]);
}
// agg[dst] += xw[src] * norm[e]; thread = edge x 8 features (16B gather,
// 4 packed-f16 atomics -> global_atomic_pk_add_f16)
__global__ void gcn_scatter_kernel(const __half2* __restrict__ xw2,
                                   const long long* __restrict__ src,
                                   const long long* __restrict__ dst,
                                   const float* __restrict__ norm,
                                   __half2* __restrict__ agg2) {
  size_t i = (size_t)blockIdx.x * blockDim.x + threadIdx.x;
  if (i >= (size_t)SC_E * 8) return;
  size_t e = i >> 3;
  int q0 = (int)(i & 7) * 4;               // half2 index within 32-pair row
  const long long s = src[e], d = dst[e];
  const __half2 nv2 = __float2half2_rn(norm[e]);
  const __half2* sp = xw2 + (size_t)s * 32 + q0;
  __half2* ap = agg2 + (size_t)d * 32 + q0;
#pragma unroll
  for (int j = 0; j < 4; ++j)
    atomicAdd(ap + j, __hmul2(sp[j], nv2));
}

// =====================================================================
extern "C" void kernel_launch(void* const* d_in, const int* in_sizes, int n_in,
                              void* d_out, int out_size, void* d_ws, size_t ws_size,
                              hipStream_t stream) {
  const float*     syn  = (const float*)d_in[0];
  const long long* ei   = (const long long*)d_in[1];   // [2, E] int64
  const float*     embW = (const float*)d_in[3];
  const float*     embB = (const float*)d_in[4];
  const float*     gcnW = (const float*)d_in[5];       // [I,64,64]
  const float*     gcnB = (const float*)d_in[6];       // [I,64]
  const float*     rW1  = (const float*)d_in[7];       // [I,64,256]
  const float*     rB1  = (const float*)d_in[8];
  const float*     rW2  = (const float*)d_in[9];       // [I,256,64]
  const float*     rB2  = (const float*)d_in[10];
  const float*     hW1  = (const float*)d_in[11];      // [I,512,256]
  const float*     hB1  = (const float*)d_in[12];
  const float*     hW2  = (const float*)d_in[13];      // [I,256,128]
  const float*     hB2  = (const float*)d_in[14];
  const float*     hW3  = (const float*)d_in[15];      // [I,128,9]
  const float*     hB3  = (const float*)d_in[16];
  float* out = (float*)d_out;

  const long long* e_src = ei;
  const long long* e_dst = ei + SC_E;

  // ---- workspace carve-up ----
  char* ws = (char*)d_ws;
  float*    dinv = (float*)ws;    ws += (size_t)SC_N * sizeof(float);
  float*    norm = (float*)ws;    ws += (size_t)SC_E * sizeof(float);
  _Float16* h    = (_Float16*)ws; ws += (size_t)SC_N * SC_F * sizeof(_Float16);
  _Float16* xw   = (_Float16*)ws; ws += (size_t)SC_N * SC_F * sizeof(_Float16);
  const int P_GCN = (SC_F / 32) * 4 * 512;          // 4096
  const int P_RW1 = (SC_F / 32) * 16 * 512;         // 16384
  const int P_RW2 = (SC_H / 32) * 4 * 512;          // 16384
  const int P_HW1 = ((SC_S * SC_F) / 32) * 16 * 512;// 131072
  const int P_HW2 = (SC_H / 32) * 8 * 512;          // 32768
  const int P_HW3 = ((SC_H / 2) / 32) * 1 * 512;    // 2048
  _Float16* p_gcn = (_Float16*)ws; ws += (size_t)SC_ITERS * P_GCN * 2;
  _Float16* p_rw1 = (_Float16*)ws; ws += (size_t)SC_ITERS * P_RW1 * 2;
  _Float16* p_rw2 = (_Float16*)ws; ws += (size_t)SC_ITERS * P_RW2 * 2;
  _Float16* p_hw1 = (_Float16*)ws; ws += (size_t)SC_ITERS * P_HW1 * 2;
  _Float16* p_hw2 = (_Float16*)ws; ws += (size_t)SC_ITERS * P_HW2 * 2;
  _Float16* p_hw3 = (_Float16*)ws; ws += (size_t)SC_ITERS * P_HW3 * 2;
  // head temporaries reuse xw (free during head phase): t1+t2 = 25.2M <= 33.6M f16
  _Float16* t1 = xw;                               // [B,256]
  _Float16* t2 = xw + (size_t)SC_B * SC_H;         // [B,128]

  const int TB = 256;
  const size_t NF = (size_t)SC_N * SC_F;

  // ---- pack all weights once (tiny) ----
  for (int it = 0; it < SC_ITERS; ++it) {
    pack_w_kernel<<<(P_GCN + TB - 1) / TB, TB, 0, stream>>>(
        gcnW + (size_t)it * SC_F * SC_F, p_gcn + (size_t)it * P_GCN, SC_F, SC_F, 4);
    pack_w_kernel<<<(P_RW1 + TB - 1) / TB, TB, 0, stream>>>(
        rW1 + (size_t)it * SC_F * SC_H, p_rw1 + (size_t)it * P_RW1, SC_F, SC_H, 16);
    pack_w_kernel<<<(P_RW2 + TB - 1) / TB, TB, 0, stream>>>(
        rW2 + (size_t)it * SC_H * SC_F, p_rw2 + (size_t)it * P_RW2, SC_H, SC_F, 4);
    pack_w_kernel<<<(P_HW1 + TB - 1) / TB, TB, 0, stream>>>(
        hW1 + (size_t)it * (SC_S * SC_F) * SC_H, p_hw1 + (size_t)it * P_HW1,
        SC_S * SC_F, SC_H, 16);
    pack_w_kernel<<<(P_HW2 + TB - 1) / TB, TB, 0, stream>>>(
        hW2 + (size_t)it * SC_H * (SC_H / 2), p_hw2 + (size_t)it * P_HW2,
        SC_H, SC_H / 2, 8);
    pack_w_kernel<<<(P_HW3 + TB - 1) / TB, TB, 0, stream>>>(
        hW3 + (size_t)it * (SC_H / 2) * SC_Q, p_hw3 + (size_t)it * P_HW3,
        SC_H / 2, SC_Q, 1);
  }

  // ---- degrees & edge norms (deterministic: +1.0f adds are exact) ----
  init_deg_kernel<<<(SC_N + TB - 1) / TB, TB, 0, stream>>>(dinv);
  deg_accum_kernel<<<(SC_E + TB - 1) / TB, TB, 0, stream>>>(e_dst, dinv);
  deg_finalize_kernel<<<(SC_N + TB - 1) / TB, TB, 0, stream>>>(dinv);
  edge_norm_kernel<<<(SC_E + TB - 1) / TB, TB, 0, stream>>>(e_src, e_dst, dinv, norm);

  // ---- embed (f16 activations) ----
  embed_kernel<<<(unsigned)((NF + TB - 1) / TB), TB, 0, stream>>>(syn, embW, embB, h);

  for (int it = 0; it < SC_ITERS; ++it) {
    // 1) xw = h @ gcnW[it]   (f16 in, f16 out)
    gemm_wmma_kernel<4, 0, false, true><<<SC_N / 128, 256, 0, stream>>>(
        h, p_gcn + (size_t)it * P_GCN, nullptr, xw, SC_N, SC_F, SC_F, 0, 4);

    // 2) h = xw*dinv^2 + bias (self loop), then edge scatter (pk f16 atomics)
    gcn_selfloop_kernel<<<(unsigned)(((size_t)SC_N * 32 + TB - 1) / TB), TB, 0, stream>>>(
        (const __half2*)xw, dinv, gcnB + (size_t)it * SC_F, (__half2*)h);
    gcn_scatter_kernel<<<(unsigned)(((size_t)SC_E * 8 + TB - 1) / TB), TB, 0, stream>>>(
        (const __half2*)xw, e_src, e_dst, norm, (__half2*)h);

    // 3) fused refine MLP with folded relu, in place on h
    refine_fused_kernel<<<SC_N / 64, 128, 0, stream>>>(
        h, p_rw1 + (size_t)it * P_RW1, rB1 + (size_t)it * SC_H,
        p_rw2 + (size_t)it * P_RW2, rB2 + (size_t)it * SC_F);

    // 4) head: [B,512] -> 256 -> 128 -> 9 (sigmoid, f32 out)
    //    first layer split into two NT=8 column-half passes
    gemm_wmma_kernel<8, 1, false, true><<<SC_B / 128, 256, 0, stream>>>(
        h, p_hw1 + (size_t)it * P_HW1, hB1 + (size_t)it * SC_H,
        t1, SC_B, SC_S * SC_F, SC_H, 0, 16);
    gemm_wmma_kernel<8, 1, false, true><<<SC_B / 128, 256, 0, stream>>>(
        h, p_hw1 + (size_t)it * P_HW1, hB1 + (size_t)it * SC_H,
        t1, SC_B, SC_S * SC_F, SC_H, 8, 16);
    gemm_wmma_kernel<8, 1, false, true><<<SC_B / 128, 256, 0, stream>>>(
        t1, p_hw2 + (size_t)it * P_HW2, hB2 + (size_t)it * (SC_H / 2),
        t2, SC_B, SC_H, SC_H / 2, 0, 8);
    gemm_wmma_kernel<1, 2, true, false><<<SC_B / 128, 256, 0, stream>>>(
        t2, p_hw3 + (size_t)it * P_HW3, hB3 + (size_t)it * SC_Q,
        out + (size_t)it * SC_B * SC_Q, SC_B, SC_H / 2, SC_Q, 0, 1);
  }
}
